// GraphLabelPropagation_17514876634035
// MI455X (gfx1250) — matrix operand order repaired and checked by reading, not compiled, not run on hardware
//
#include <hip/hip_runtime.h>
#include <hip/hip_bf16.h>
#include <stdint.h>

// ---------------------------------------------------------------------------
// Graph label propagation on MI455X (gfx1250, wave32, WMMA + TDM).
// Heavy phase: Xn @ Xn^T (2*N^2*D ~ 1e11 FLOP) via V_WMMA_F32_16X16X32_F16.
// Block stages a 32x128 f16 A-tile in LDS via tensor_load_to_lds (TDM);
// each wave computes a 32x64 output patch (32 WMMAs, full-EXEC, no
// divergence in the compute loop: overhang tiles are clamped, store-guarded).
// ---------------------------------------------------------------------------

#define N_SAMP      20000
#define D_FEAT      128
#define K_NN        50
#define NUM_CLS     10
#define ALPHA_LP    0.99f
#define MAX_ITER    20
#define ROW_CHUNK   160                      // rows of sims materialized per pass
#define COL_TILES   (N_SAMP / 16)            // 1250
#define EPS_CG      1e-30f
#define BM_WORDS    ((N_SAMP + 31) / 32)     // 625

typedef __attribute__((ext_vector_type(16))) _Float16 v16h;
typedef __attribute__((ext_vector_type(8)))  float    v8f;
typedef unsigned int v4u __attribute__((ext_vector_type(4)));
typedef int v8i __attribute__((ext_vector_type(8)));
typedef int v4i __attribute__((ext_vector_type(4)));

#if defined(__has_builtin)
#if __has_builtin(__builtin_amdgcn_tensor_load_to_lds) && \
    __has_builtin(__builtin_amdgcn_s_wait_tensorcnt)
#define HAVE_TDM 1
#endif
#endif

// ---------------- Phase 0: row L2-normalize, f32 -> f16 --------------------
__global__ void k_normalize(const float* __restrict__ X, _Float16* __restrict__ Xh) {
    int wave = threadIdx.x >> 5;
    int lane = threadIdx.x & 31;
    int row  = blockIdx.x * (blockDim.x >> 5) + wave;
    if (row >= N_SAMP) return;                       // wave-uniform
    const float4 v = ((const float4*)(X + (size_t)row * D_FEAT))[lane]; // 4 f32/lane
    float ss = v.x * v.x + v.y * v.y + v.z * v.z + v.w * v.w;
    #pragma unroll
    for (int m = 16; m >= 1; m >>= 1) ss += __shfl_xor(ss, m, 32);
    float sc = rsqrtf(fmaxf(ss, 1e-30f));
    _Float16* o = Xh + (size_t)row * D_FEAT + lane * 4;
    o[0] = (_Float16)(v.x * sc);
    o[1] = (_Float16)(v.y * sc);
    o[2] = (_Float16)(v.z * sc);
    o[3] = (_Float16)(v.w * sc);
}

// ---------------- Phase 1: WMMA GEMM, sims[rlocal][c] ----------------------
// Block = 8 waves. A tile = rows [ra, ra+32) staged in LDS (contiguous 8KB).
// Wave computes 2 row-tiles x 4 col-tiles of 16x16 each.
// A frag (16x32 f16): lane L<16 -> row +L, K in {0..7,16..23}; L>=16 -> K+8.
// B frag (32x16 f16): lane L -> col cb+(L&15), K in {0..7,16..23}; L>=16 -> K+8.
__global__ void k_sims(const _Float16* __restrict__ Xh, float* __restrict__ sims, int rb) {
    __shared__ _Float16 lA[32 * D_FEAT];             // 8 KB staged A tile
    int wave  = threadIdx.x >> 5;
    int lane  = threadIdx.x & 31;
    int l15   = lane & 15;
    int khalf = lane >> 4;
    int ra    = rb + blockIdx.y * 32;                // global row base of A tile

#ifdef HAVE_TDM
    if (wave == 0) {
        // 1-D TDM descriptor: 4096 contiguous 2-byte elements -> LDS.
        uint64_t ga = (uint64_t)(uintptr_t)(Xh + (size_t)ra * D_FEAT);
        uint32_t lo = (uint32_t)(uintptr_t)(void*)lA;   // LDS byte offset
        const unsigned ne = 32u * D_FEAT;               // 4096 elements
        v4u g0; v8i g1;
        v4i g2 = {0, 0, 0, 0}, g3 = {0, 0, 0, 0};
        v8i g4 = {0, 0, 0, 0, 0, 0, 0, 0};
        g0[0] = 1u;                                     // count=1, gather off
        g0[1] = lo;                                     // lds_addr
        g0[2] = (uint32_t)ga;                           // global_addr[31:0]
        g0[3] = (uint32_t)(ga >> 32) | (2u << 30);      // global_addr[56:32] | type=2
        g1[0] = (int)(1u << 16);                        // data_size = 1 (2 bytes)
        g1[1] = (int)(ne << 16);                        // tensor_dim0[15:0]
        g1[2] = (int)(1u << 16);                        // dim0 hi=0 | tensor_dim1=1
        g1[3] = (int)(ne << 16);                        // tile_dim0 = 4096
        g1[4] = 0;                                      // tile_dim1/2 unused
        g1[5] = (int)ne;                                // tensor_dim0_stride lo
        g1[6] = 0; g1[7] = 0;
        __builtin_amdgcn_tensor_load_to_lds(g0, g1, g2, g3, g4, 0);  // 6-arg form
        __builtin_amdgcn_s_wait_tensorcnt(0);
    }
#else
    {   // cooperative fallback: 8 KB via b128 copies
        const uint4* src = (const uint4*)(Xh + (size_t)ra * D_FEAT);
        uint4* dst = (uint4*)lA;
        for (int i = threadIdx.x; i < 32 * D_FEAT * 2 / 16; i += 256) dst[i] = src[i];
    }
#endif
    __syncthreads();

    // Scalarized first col-tile index: guards become s_cmp/s_cbranch, and the
    // WMMA loop runs unconditionally with full EXEC.
    int cg0 = __builtin_amdgcn_readfirstlane((blockIdx.x * 8 + wave) * 4);
    if (cg0 >= COL_TILES) return;                    // scalar branch, after barrier

    const uint32_t* la  = (const uint32_t*)lA;
    const uint32_t* pa0 = la + (size_t)l15 * (D_FEAT / 2);
    const uint32_t* pa1 = la + (size_t)(16 + l15) * (D_FEAT / 2);

    // Clamp overhang tiles (last wave-group only): redundant compute, no OOB.
    int ct[4];
    #pragma unroll
    for (int j = 0; j < 4; ++j) {
        int jt = cg0 + j;
        ct[j] = (jt < COL_TILES) ? jt : (COL_TILES - 1);
    }

    v8f acc[2][4] = {};
    #pragma unroll
    for (int k0 = 0; k0 < D_FEAT; k0 += 32) {
        int base = k0 >> 1;                          // u32 index into a row
        union { v16h h; uint32_t u[8]; } A0, A1;
        #pragma unroll
        for (int j = 0; j < 4; ++j) {                // two 16B runs per lane (LDS)
            A0.u[j]     = pa0[base + 4 * khalf + j];
            A0.u[4 + j] = pa0[base + 8 + 4 * khalf + j];
            A1.u[j]     = pa1[base + 4 * khalf + j];
            A1.u[4 + j] = pa1[base + 8 + 4 * khalf + j];
        }
        #pragma unroll
        for (int j = 0; j < 4; ++j) {
            const uint32_t* pbj =
                (const uint32_t*)(Xh + (size_t)(ct[j] * 16 + l15) * D_FEAT);
            union { v16h h; uint32_t u[8]; } B;
            #pragma unroll
            for (int q = 0; q < 8; ++q)              // two 16B runs per lane
                B.u[q] = pbj[base + 4 * khalf + q + (q & 4)];
            acc[0][j] = __builtin_amdgcn_wmma_f32_16x16x32_f16(
                false, A0.h, false, B.h, (short)0, acc[0][j], false, false);
            acc[1][j] = __builtin_amdgcn_wmma_f32_16x16x32_f16(
                false, A1.h, false, B.h, (short)0, acc[1][j], false, false);
        }
    }
    // C layout: VGPR r, lanes 0-15 -> M=r, lanes 16-31 -> M=r+8; N = lane&15.
    #pragma unroll
    for (int j = 0; j < 4; ++j) {
        if (cg0 + j >= COL_TILES) continue;          // scalar store guard
        int col = (cg0 + j) * 16 + l15;
        #pragma unroll
        for (int rt = 0; rt < 2; ++rt) {
            size_t rloc0 = (size_t)(blockIdx.y * 32 + rt * 16 + 8 * khalf);
            float* o = sims + rloc0 * N_SAMP + col;
            #pragma unroll
            for (int r = 0; r < 8; ++r) o[(size_t)r * N_SAMP] = acc[rt][j][r];
        }
    }
}

// ---------------- Phase 2: per-row top-51 (drop self), D^3 -----------------
#define TK_T 256
__global__ void k_topk(const float* __restrict__ sims, int* __restrict__ I,
                       float* __restrict__ V, int rb) {
    __shared__ uint32_t bm[BM_WORDS];
    __shared__ float    redv[TK_T];
    __shared__ int      redi[TK_T];
    __shared__ float    topv[K_NN + 1];
    __shared__ int      topi[K_NN + 1];
    int t = threadIdx.x;
    int rlocal = blockIdx.x;
    int row = rb + rlocal;
    const float* srow = sims + (size_t)rlocal * N_SAMP;

    for (int i = t; i < BM_WORDS; i += TK_T) bm[i] = 0u;
    __syncthreads();

    float bestv = -1e30f; int besti = -1;
    for (int c = t; c < N_SAMP; c += TK_T) {
        float v = srow[c];
        if (v > bestv) { bestv = v; besti = c; }
    }
    for (int j = 0; j <= K_NN; ++j) {                // 51 extractions
        redv[t] = bestv; redi[t] = besti;
        __syncthreads();
        for (int s = TK_T / 2; s > 0; s >>= 1) {
            if (t < s) {
                float ov = redv[t + s]; int oi = redi[t + s];
                if (ov > redv[t] || (ov == redv[t] && oi >= 0 && oi < redi[t])) {
                    redv[t] = ov; redi[t] = oi;
                }
            }
            __syncthreads();
        }
        if (t == 0) {
            topv[j] = redv[0]; topi[j] = redi[0];
            int w = redi[0];
            if (w >= 0) bm[w >> 5] |= (1u << (w & 31));
        }
        __syncthreads();
        int widx = topi[j];
        if (widx >= 0 && t == (widx & (TK_T - 1))) { // winner re-derives its local max
            bestv = -1e30f; besti = -1;
            for (int c = t; c < N_SAMP; c += TK_T) {
                if (bm[c >> 5] & (1u << (c & 31))) continue;
                float v = srow[c];
                if (v > bestv) { bestv = v; besti = c; }
            }
        }
        __syncthreads();
    }
    if (t >= 1 && t <= K_NN) {                       // drop slot 0 (self), D = d^3
        float d = topv[t];
        I[(size_t)row * K_NN + (t - 1)] = topi[t];
        V[(size_t)row * K_NN + (t - 1)] = d * d * d;
    }
}

// ---------------- Graph build + CG kernels ---------------------------------
__global__ void k_zero(float* p, int n) {
    int i = blockIdx.x * blockDim.x + threadIdx.x;
    if (i < n) p[i] = 0.0f;
}

__global__ void k_degree(const int* __restrict__ I, const float* __restrict__ V,
                         float* __restrict__ S) {
    int e = blockIdx.x * blockDim.x + threadIdx.x;
    if (e >= N_SAMP * K_NN) return;
    int r = e / K_NN, c = I[e];
    if (r == c) return;                              // diagonal removed
    float v = V[e];
    atomicAdd(&S[r], v);
    atomicAdd(&S[c], v);
}

__global__ void k_dinv(const float* __restrict__ S, float* __restrict__ dinv) {
    int i = blockIdx.x * blockDim.x + threadIdx.x;
    if (i >= N_SAMP) return;
    float s = S[i]; if (s == 0.0f) s = 1.0f;
    dinv[i] = rsqrtf(s);
}

__global__ void k_counts(const int* __restrict__ labels, const int* __restrict__ mask,
                         float* __restrict__ counts) {
    int i = blockIdx.x * blockDim.x + threadIdx.x;
    if (i >= N_SAMP) return;
    if (mask[i] != 0) atomicAdd(&counts[labels[i]], 1.0f);
}

__global__ void k_initY(const int* __restrict__ labels, const int* __restrict__ mask,
                        const float* __restrict__ counts, float* __restrict__ x,
                        float* __restrict__ r, float* __restrict__ p) {
    int i = blockIdx.x * blockDim.x + threadIdx.x;
    if (i >= N_SAMP) return;
    int lab = labels[i], m = mask[i];
    #pragma unroll
    for (int c = 0; c < NUM_CLS; ++c) {
        float cnt = counts[c]; if (cnt == 0.0f) cnt = 1.0f;
        float y = (m != 0 && lab == c) ? (1.0f / cnt) : 0.0f;
        x[(size_t)i * NUM_CLS + c] = 0.0f;
        r[(size_t)i * NUM_CLS + c] = y;
        p[(size_t)i * NUM_CLS + c] = y;
    }
}

__global__ void k_coldot(const float* __restrict__ a, const float* __restrict__ b,
                         float* __restrict__ out10) {
    __shared__ float s[NUM_CLS];
    if (threadIdx.x < NUM_CLS) s[threadIdx.x] = 0.0f;
    __syncthreads();
    int i = blockIdx.x * blockDim.x + threadIdx.x;
    if (i < N_SAMP) {
        #pragma unroll
        for (int c = 0; c < NUM_CLS; ++c)
            atomicAdd(&s[c], a[(size_t)i * NUM_CLS + c] * b[(size_t)i * NUM_CLS + c]);
    }
    __syncthreads();
    if (threadIdx.x < NUM_CLS) atomicAdd(&out10[threadIdx.x], s[threadIdx.x]);
}

__global__ void k_spmv(const int* __restrict__ I, const float* __restrict__ V,
                       const float* __restrict__ dinv, const float* __restrict__ p,
                       float* __restrict__ wp) {
    int e = blockIdx.x * blockDim.x + threadIdx.x;
    if (e >= N_SAMP * K_NN) return;
    int r = e / K_NN, c = I[e];
    if (r == c) return;
    float w = V[e] * dinv[r] * dinv[c];              // symmetric normalization
    #pragma unroll
    for (int cl = 0; cl < NUM_CLS; ++cl) {
        atomicAdd(&wp[(size_t)r * NUM_CLS + cl], w * p[(size_t)c * NUM_CLS + cl]);
        atomicAdd(&wp[(size_t)c * NUM_CLS + cl], w * p[(size_t)r * NUM_CLS + cl]);
    }
}

__global__ void k_ap(const float* __restrict__ p, const float* __restrict__ wp,
                     float* __restrict__ Ap, float* __restrict__ pAp) {
    __shared__ float s[NUM_CLS];
    if (threadIdx.x < NUM_CLS) s[threadIdx.x] = 0.0f;
    __syncthreads();
    int i = blockIdx.x * blockDim.x + threadIdx.x;
    if (i < N_SAMP) {
        #pragma unroll
        for (int c = 0; c < NUM_CLS; ++c) {
            size_t k = (size_t)i * NUM_CLS + c;
            float ap = p[k] - ALPHA_LP * wp[k];      // A = I - alpha * Wn
            Ap[k] = ap;
            atomicAdd(&s[c], p[k] * ap);
        }
    }
    __syncthreads();
    if (threadIdx.x < NUM_CLS) atomicAdd(&pAp[threadIdx.x], s[threadIdx.x]);
}

__global__ void k_alpha(const float* __restrict__ rs, const float* __restrict__ pAp,
                        float* __restrict__ a) {
    int c = threadIdx.x;
    if (c < NUM_CLS) a[c] = rs[c] / (pAp[c] + EPS_CG);
}

__global__ void k_xr(const float* __restrict__ a, const float* __restrict__ p,
                     const float* __restrict__ Ap, float* __restrict__ x,
                     float* __restrict__ r, float* __restrict__ rsnew) {
    __shared__ float s[NUM_CLS];
    if (threadIdx.x < NUM_CLS) s[threadIdx.x] = 0.0f;
    __syncthreads();
    int i = blockIdx.x * blockDim.x + threadIdx.x;
    if (i < N_SAMP) {
        #pragma unroll
        for (int c = 0; c < NUM_CLS; ++c) {
            size_t k = (size_t)i * NUM_CLS + c;
            x[k] = x[k] + a[c] * p[k];
            float rv = r[k] - a[c] * Ap[k];
            r[k] = rv;
            atomicAdd(&s[c], rv * rv);
        }
    }
    __syncthreads();
    if (threadIdx.x < NUM_CLS) atomicAdd(&rsnew[threadIdx.x], s[threadIdx.x]);
}

__global__ void k_beta(float* __restrict__ rs, const float* __restrict__ rsnew,
                       float* __restrict__ beta) {
    int c = threadIdx.x;
    if (c < NUM_CLS) { beta[c] = rsnew[c] / (rs[c] + EPS_CG); rs[c] = rsnew[c]; }
}

__global__ void k_p(const float* __restrict__ beta, const float* __restrict__ r,
                    float* __restrict__ p) {
    int i = blockIdx.x * blockDim.x + threadIdx.x;
    if (i >= N_SAMP) return;
    #pragma unroll
    for (int c = 0; c < NUM_CLS; ++c) {
        size_t k = (size_t)i * NUM_CLS + c;
        p[k] = r[k] + beta[c] * p[k];
    }
}

// ---------------- Final: argmax, accuracy, label write-back ----------------
__global__ void k_final(const float* __restrict__ x, const int* __restrict__ labels,
                        const int* __restrict__ mask, float* __restrict__ out,
                        float* __restrict__ accum) {
    __shared__ float s;
    if (threadIdx.x == 0) s = 0.0f;
    __syncthreads();
    int i = blockIdx.x * blockDim.x + threadIdx.x;
    float local = 0.0f;
    if (i < N_SAMP) {
        float bv = fmaxf(x[(size_t)i * NUM_CLS], 0.0f);
        int bi = 0;
        #pragma unroll
        for (int c = 1; c < NUM_CLS; ++c) {
            float z = fmaxf(x[(size_t)i * NUM_CLS + c], 0.0f);
            if (z > bv) { bv = z; bi = c; }
        }
        int lab = labels[i];
        local = (bi == lab) ? 1.0f : 0.0f;           // acc uses pre-overwrite argmax
        int pl = (mask[i] != 0) ? lab : bi;
        out[i] = (float)pl;
    }
    atomicAdd(&s, local);
    __syncthreads();
    if (threadIdx.x == 0) atomicAdd(accum, s);
}

__global__ void k_acc(const float* __restrict__ accum, float* __restrict__ out) {
    if (threadIdx.x == 0) out[N_SAMP] = accum[0] / (float)N_SAMP;
}

// ---------------------------------------------------------------------------
extern "C" void kernel_launch(void* const* d_in, const int* in_sizes, int n_in,
                              void* d_out, int out_size, void* d_ws, size_t ws_size,
                              hipStream_t stream) {
    const float* X      = (const float*)d_in[0];
    const int*   labels = (const int*)  d_in[1];
    const int*   mask   = (const int*)  d_in[2];
    float*       out    = (float*)      d_out;

    // --- workspace layout (256B-aligned slabs) ---
    char* w = (char*)d_ws;
    size_t off = 0;
    auto alloc = [&](size_t bytes) -> void* {
        void* p = w + off;
        off = (off + bytes + 255) & ~(size_t)255;
        return p;
    };
    _Float16* Xh    = (_Float16*)alloc((size_t)N_SAMP * D_FEAT * 2);
    float*    sims  = (float*)   alloc((size_t)ROW_CHUNK * N_SAMP * 4);
    int*      Iknn  = (int*)     alloc((size_t)N_SAMP * K_NN * 4);
    float*    Vknn  = (float*)   alloc((size_t)N_SAMP * K_NN * 4);
    float*    S     = (float*)   alloc((size_t)N_SAMP * 4);
    float*    dinvb = (float*)   alloc((size_t)N_SAMP * 4);
    float*    xb    = (float*)   alloc((size_t)N_SAMP * NUM_CLS * 4);
    float*    rbuf  = (float*)   alloc((size_t)N_SAMP * NUM_CLS * 4);
    float*    pbuf  = (float*)   alloc((size_t)N_SAMP * NUM_CLS * 4);
    float*    Apb   = (float*)   alloc((size_t)N_SAMP * NUM_CLS * 4);
    float*    wpb   = (float*)   alloc((size_t)N_SAMP * NUM_CLS * 4);
    float*    scal  = (float*)   alloc(256 * 4);
    float* counts = scal +  0;   // 10
    float* rs     = scal + 16;   // 10
    float* pAp    = scal + 32;   // 10
    float* aco    = scal + 48;   // 10
    float* rsnew  = scal + 64;   // 10
    float* betac  = scal + 80;   // 10
    float* accum  = scal + 96;   // 1
    (void)in_sizes; (void)n_in; (void)out_size; (void)ws_size;

    const int NB   = (N_SAMP + 255) / 256;                     // 79
    const int NEB  = (N_SAMP * K_NN + 255) / 256;              // 3907
    const int NCB  = (N_SAMP * NUM_CLS + 255) / 256;           // 782 (full wp zero)

    // Phase 0: normalize + f16 convert (wave per row, 4 waves/block)
    k_normalize<<<dim3((N_SAMP + 3) / 4), 128, 0, stream>>>(X, Xh);

    // Phase 1+2: chunked WMMA GEMM + top-k (125 chunks of 160 rows)
    // k_sims grid: x covers 32 col-tiles/block (8 waves x 4 tiles), y = row pairs.
    for (int rbase = 0; rbase < N_SAMP; rbase += ROW_CHUNK) {
        k_sims<<<dim3((COL_TILES + 31) / 32, ROW_CHUNK / 32), 256, 0, stream>>>(Xh, sims, rbase);
        k_topk<<<dim3(ROW_CHUNK), TK_T, 0, stream>>>(sims, Iknn, Vknn, rbase);
    }

    // Graph normalization
    k_zero<<<NB, 256, 0, stream>>>(S, N_SAMP);
    k_degree<<<NEB, 256, 0, stream>>>(Iknn, Vknn, S);
    k_dinv<<<NB, 256, 0, stream>>>(S, dinvb);

    // RHS: Y = onehot * mask / counts ; x=0, r=p=Y, rs = sum(Y*Y)
    k_zero<<<1, 256, 0, stream>>>(scal, 256);
    k_counts<<<NB, 256, 0, stream>>>(labels, mask, counts);
    k_initY<<<NB, 256, 0, stream>>>(labels, mask, counts, xb, rbuf, pbuf);
    k_coldot<<<NB, 256, 0, stream>>>(rbuf, rbuf, rs);

    // CG, 20 fixed iterations
    for (int it = 0; it < MAX_ITER; ++it) {
        k_zero<<<NCB, 256, 0, stream>>>(wpb, N_SAMP * NUM_CLS);
        k_zero<<<1, 32, 0, stream>>>(pAp, NUM_CLS);
        k_spmv<<<NEB, 256, 0, stream>>>(Iknn, Vknn, dinvb, pbuf, wpb);
        k_ap<<<NB, 256, 0, stream>>>(pbuf, wpb, Apb, pAp);
        k_alpha<<<1, 32, 0, stream>>>(rs, pAp, aco);
        k_zero<<<1, 32, 0, stream>>>(rsnew, NUM_CLS);
        k_xr<<<NB, 256, 0, stream>>>(aco, pbuf, Apb, xb, rbuf, rsnew);
        k_beta<<<1, 32, 0, stream>>>(rs, rsnew, betac);
        k_p<<<NB, 256, 0, stream>>>(betac, rbuf, pbuf);
    }

    // Final: argmax -> accuracy -> label override; out[N] = acc
    k_zero<<<1, 32, 0, stream>>>(accum, 1);
    k_final<<<NB, 256, 0, stream>>>(xb, labels, mask, out, accum);
    k_acc<<<1, 32, 0, stream>>>(accum, out);
}
